// AttentionLayer_58308476010683
// MI455X (gfx1250) — matrix operand order repaired and checked by reading, not compiled, not run on hardware
//
#include <hip/hip_runtime.h>
#include <hip/hip_bf16.h>

typedef __bf16 bf16_t;
typedef __attribute__((ext_vector_type(16))) __bf16 v16bf;
typedef __attribute__((ext_vector_type(8)))  __bf16 v8bf;
typedef __attribute__((ext_vector_type(8)))  float  v8f;
typedef __attribute__((ext_vector_type(4)))  float  v4f;
typedef __attribute__((ext_vector_type(4)))  unsigned v4u;

#define IN_F   512
#define OUT_F  512
#define NROWS  8192

// float -> bf16 bits, round-to-nearest-even
__device__ __forceinline__ unsigned short bfbits(float f) {
  unsigned u = __builtin_bit_cast(unsigned, f);
  u = (u + 0x7FFFu + ((u >> 16) & 1u)) >> 16;
  return (unsigned short)u;
}
__device__ __forceinline__ bf16_t f2bf(float f) {
  return __builtin_bit_cast(bf16_t, bfbits(f));
}
__device__ __forceinline__ unsigned pack2bf(float a, float b) {
  return (unsigned)bfbits(a) | ((unsigned)bfbits(b) << 16);
}

// CDNA5 async copy: global -> LDS, 16B per lane, tracked by ASYNCcnt
__device__ __forceinline__ void async_copy_b128(void* lds, const void* g) {
  unsigned l = (unsigned)(uintptr_t)lds;  // low 32 bits of generic ptr = LDS byte addr
  asm volatile("global_load_async_to_lds_b128 %0, %1, off"
               :: "v"(l), "v"(g) : "memory");
}
__device__ __forceinline__ void wait_asynccnt0() {
  asm volatile("s_wait_asynccnt 0x0" ::: "memory");
}

// A-fragment (16-bit A 16x32): elems 0..15 at p[0..15]
__device__ __forceinline__ v16bf load_frag_2x8(const bf16_t* p) {
  v8bf lo = *(const v8bf*)(p);
  v8bf hi = *(const v8bf*)(p + 8);
  v16bf r;
#pragma unroll
  for (int e = 0; e < 8; ++e) { r[e] = lo[e]; r[8 + e] = hi[e]; }
  return r;
}

// B-fragment: 16 contiguous K-values per lane
__device__ __forceinline__ v16bf load_frag_16(const bf16_t* p) {
  v8bf lo = *(const v8bf*)(p);
  v8bf hi = *(const v8bf*)(p + 8);
  v16bf r;
#pragma unroll
  for (int e = 0; e < 8; ++e) { r[e] = lo[e]; r[8 + e] = hi[e]; }
  return r;
}

__device__ __forceinline__ v8f wmma_bf16(v16bf a, v16bf b, v8f c) {
  return __builtin_amdgcn_wmma_f32_16x16x32_bf16(false, a, false, b, (short)0, c,
                                                 false, false);
}

// ---------------------------------------------------------------------------
// T[C][R] = bf16( S[R][C] ^T ), LDS-tiled 64x64. grid = (R/64, C/64)
// ---------------------------------------------------------------------------
__global__ void __launch_bounds__(256)
k_transpose_bf16(const float* __restrict__ S, bf16_t* __restrict__ T,
                 int C, int R) {
  __shared__ float tile[64][65];
  const int r0 = blockIdx.x * 64;
  const int c0 = blockIdx.y * 64;
  const int tid = threadIdx.x;
  for (int l = tid; l < 64 * 64; l += 256) {
    int i = l >> 6, j = l & 63;
    tile[i][j] = S[(size_t)(r0 + i) * C + c0 + j];
  }
  __syncthreads();
  for (int l = tid; l < 64 * 64; l += 256) {
    int j = l >> 6, i = l & 63;
    T[(size_t)(c0 + j) * R + (r0 + i)] = f2bf(tile[i][j]);
  }
}

// ---------------------------------------------------------------------------
// H[R,512] = bf16( F[R,512](f32) @ W ), with W given pre-transposed as
// Wt[OUT][IN] bf16 so B-fragments are contiguous global b128 loads.
// Block: 32 rows x 128 cols; 8 waves, wave w owns cols 16w..16w+15, rows 0..31.
// ---------------------------------------------------------------------------
__global__ void __launch_bounds__(256)
k_gemm_h(const float* __restrict__ F, const bf16_t* __restrict__ Wt,
         bf16_t* __restrict__ H) {
  __shared__ bf16_t A_lds[32 * IN_F];   // 32 KB
  const int tid  = threadIdx.x;
  const int lane = tid & 31, wv = tid >> 5;
  const int ml   = lane & 15;
  const int hi   = (lane & 16) ? 1 : 0;
  const int rb   = blockIdx.x * 32;
  const int nb   = blockIdx.y * 128;

  // Stage 32x512 f32 row-block (contiguous in memory) -> bf16 LDS, vectorized.
  const float* Fblk = &F[(size_t)rb * IN_F];
#pragma unroll
  for (int l = tid; l < (32 * IN_F) / 8; l += 256) {
    v4f x = *(const v4f*)(Fblk + 8 * l);
    v4f y = *(const v4f*)(Fblk + 8 * l + 4);
    v4u p = {pack2bf(x[0], x[1]), pack2bf(x[2], x[3]),
             pack2bf(y[0], y[1]), pack2bf(y[2], y[3])};
    *(v4u*)&A_lds[8 * l] = p;
  }
  __syncthreads();

  v8f acc0 = {0.f, 0.f, 0.f, 0.f, 0.f, 0.f, 0.f, 0.f};
  v8f acc1 = acc0;
  const bf16_t* wrow = &Wt[(size_t)(nb + 16 * wv + ml) * IN_F];

#pragma unroll 4
  for (int kk = 0; kk < IN_F; kk += 32) {
    v16bf a0 = load_frag_2x8(&A_lds[ml * IN_F + kk + hi * 16]);
    v16bf a1 = load_frag_2x8(&A_lds[(16 + ml) * IN_F + kk + hi * 16]);
    v16bf b  = load_frag_16(wrow + kk + hi * 16);
    acc0 = wmma_bf16(a0, b, acc0);
    acc1 = wmma_bf16(a1, b, acc1);
  }

  const int col = nb + 16 * wv + ml;
#pragma unroll
  for (int r = 0; r < 8; ++r) {
    int row0 = rb + r + hi * 8;
    H[(size_t)row0 * OUT_F + col]        = f2bf(acc0[r]);
    H[(size_t)(row0 + 16) * OUT_F + col] = f2bf(acc1[r]);
  }
}

// ---------------------------------------------------------------------------
// Fused flash-style pass. Per WG: 32 output rows x all 512 cols.
// For each 128-row chunk c of Hs/feat_s:
//   S[32,128] = Hp_blk @ Hs_chunk^T   (bf16 WMMA, f32 acc)
//   relu, sumsq accumulate, bf16 S tile -> LDS
//   Cacc[32,512] += S_tile @ feat_s_chunk   (B-frags from transposed Ts)
// Writes unnormalized C to d_out, per-WG sumsq partial to partials[].
// ---------------------------------------------------------------------------
__global__ void __launch_bounds__(256)
k_fused(const bf16_t* __restrict__ Hp, const bf16_t* __restrict__ Hs,
        const bf16_t* __restrict__ Ts, float* __restrict__ Cout,
        float* __restrict__ partials) {
  __shared__ bf16_t Hp_lds[32 * 512];  // 32 KB
  __shared__ bf16_t St[32 * 128];      // 8 KB
  __shared__ float  red[256];

  const int tid  = threadIdx.x;
  const int lane = tid & 31, wv = tid >> 5;
  const int ml   = lane & 15;
  const int hi   = (lane & 16) ? 1 : 0;
  const int rb   = blockIdx.x * 32;

  // Async DMA the (contiguous) 32KB bf16 row-block into LDS.
  {
    const char* gsrc = (const char*)&Hp[(size_t)rb * 512];
    char* ldst = (char*)Hp_lds;
#pragma unroll
    for (int it = 0; it < 8; ++it) {
      int off = (it * 256 + tid) * 16;
      async_copy_b128(ldst + off, gsrc + off);
    }
    wait_asynccnt0();
  }
  __syncthreads();

  const v8f vzero = {0.f, 0.f, 0.f, 0.f, 0.f, 0.f, 0.f, 0.f};
  v8f cacc[2][4];
#pragma unroll
  for (int a = 0; a < 2; ++a)
#pragma unroll
    for (int b = 0; b < 4; ++b) cacc[a][b] = vzero;

  float ssq = 0.f;

  for (int c = 0; c < NROWS; c += 128) {
    // ---- stage 1: S = Hp_blk @ Hs_chunk^T (wave owns S cols 16w..16w+15)
    v8f s0 = vzero, s1 = vzero;
    const bf16_t* hsrow = &Hs[(size_t)(c + 16 * wv + ml) * 512];
    if (c + 128 < NROWS) {
      __builtin_prefetch(&Hs[(size_t)(c + 128 + 16 * wv + ml) * 512], 0, 0);
      __builtin_prefetch(&Ts[(size_t)(64 * wv + ml) * NROWS + c + 128], 0, 0);
    }
#pragma unroll 4
    for (int kk = 0; kk < 512; kk += 32) {
      v16bf a0 = load_frag_2x8(&Hp_lds[ml * 512 + kk + hi * 16]);
      v16bf a1 = load_frag_2x8(&Hp_lds[(16 + ml) * 512 + kk + hi * 16]);
      v16bf b  = load_frag_16(hsrow + kk + hi * 16);
      s0 = wmma_bf16(a0, b, s0);
      s1 = wmma_bf16(a1, b, s1);
    }

    __syncthreads();  // previous iteration's stage-2 reads of St are done
    const int scol = 16 * wv + ml;
#pragma unroll
    for (int r = 0; r < 8; ++r) {
      float v0 = fmaxf(s0[r], 0.f);
      float v1 = fmaxf(s1[r], 0.f);
      ssq += v0 * v0 + v1 * v1;
      St[(r + hi * 8) * 128 + scol]      = f2bf(v0);
      St[(16 + r + hi * 8) * 128 + scol] = f2bf(v1);
    }
    __syncthreads();

    // ---- stage 2: Cacc += St @ feat_s_chunk (wave owns C cols 64w..64w+63)
#pragma unroll
    for (int k2 = 0; k2 < 128; k2 += 32) {
      v16bf a0 = load_frag_2x8(&St[ml * 128 + k2 + hi * 16]);
      v16bf a1 = load_frag_2x8(&St[(16 + ml) * 128 + k2 + hi * 16]);
#pragma unroll
      for (int t = 0; t < 4; ++t) {
        int colg = 64 * wv + 16 * t + ml;
        v16bf b = load_frag_16(&Ts[(size_t)colg * NROWS + c + k2 + hi * 16]);
        cacc[0][t] = wmma_bf16(a0, b, cacc[0][t]);
        cacc[1][t] = wmma_bf16(a1, b, cacc[1][t]);
      }
    }
  }

  // ---- write unnormalized C
#pragma unroll
  for (int t = 0; t < 4; ++t) {
    int col = 64 * wv + 16 * t + ml;
#pragma unroll
    for (int r = 0; r < 8; ++r) {
      int row0 = rb + r + hi * 8;
      Cout[(size_t)row0 * 512 + col]        = cacc[0][t][r];
      Cout[(size_t)(row0 + 16) * 512 + col] = cacc[1][t][r];
    }
  }

  // ---- sumsq partial
  red[tid] = ssq;
  __syncthreads();
  for (int s = 128; s > 0; s >>= 1) {
    if (tid < s) red[tid] += red[tid + s];
    __syncthreads();
  }
  if (tid == 0) partials[blockIdx.x] = red[0];
}

// ---------------------------------------------------------------------------
// out *= rsqrt(sum(partials[0..255]))
// ---------------------------------------------------------------------------
__global__ void __launch_bounds__(256)
k_scale(float* __restrict__ Cout, const float* __restrict__ partials) {
  __shared__ float red[256];
  const int tid = threadIdx.x;
  red[tid] = partials[tid];
  __syncthreads();
  for (int s = 128; s > 0; s >>= 1) {
    if (tid < s) red[tid] += red[tid + s];
    __syncthreads();
  }
  const float scale = rsqrtf(red[0]);
  const size_t base = (size_t)blockIdx.x * 4096;
  for (int i = tid; i < 4096; i += 256) Cout[base + i] *= scale;
}

extern "C" void kernel_launch(void* const* d_in, const int* in_sizes, int n_in,
                              void* d_out, int out_size, void* d_ws, size_t ws_size,
                              hipStream_t stream) {
  (void)in_sizes; (void)n_in; (void)out_size; (void)ws_size;
  const float* feat_p = (const float*)d_in[0];
  const float* feat_s = (const float*)d_in[1];
  const float* weight = (const float*)d_in[2];
  float* out = (float*)d_out;

  char* ws = (char*)d_ws;
  const size_t HBYTES = (size_t)NROWS * 512 * sizeof(bf16_t);  // 8 MB
  bf16_t* Hp  = (bf16_t*)(ws);
  bf16_t* Hs  = (bf16_t*)(ws + HBYTES);
  bf16_t* Ts  = (bf16_t*)(ws + 2 * HBYTES);
  bf16_t* WtG = (bf16_t*)(ws + 3 * HBYTES);                    // 512 KB
  float*  partials = (float*)(ws + 3 * HBYTES + (size_t)OUT_F * IN_F * 2);

  dim3 blk(256);
  // W^T (bf16) once: B-operand for both H GEMMs
  k_transpose_bf16<<<dim3(IN_F / 64, OUT_F / 64), blk, 0, stream>>>(weight, WtG,
                                                                    OUT_F, IN_F);
  k_gemm_h<<<dim3(NROWS / 32, OUT_F / 128), blk, 0, stream>>>(feat_p, WtG, Hp);
  k_gemm_h<<<dim3(NROWS / 32, OUT_F / 128), blk, 0, stream>>>(feat_s, WtG, Hs);
  k_transpose_bf16<<<dim3(NROWS / 64, IN_F / 64), blk, 0, stream>>>(feat_s, Ts,
                                                                    IN_F, NROWS);
  k_fused<<<dim3(NROWS / 32), blk, 0, stream>>>(Hp, Hs, Ts, out, partials);
  k_scale<<<dim3((NROWS * 512) / 4096), blk, 0, stream>>>(out, partials);
}
